// PassGNN_49555332661729
// MI455X (gfx1250) — matrix operand order repaired and checked by reading, not compiled, not run on hardware
//
#include <hip/hip_runtime.h>
#include <hip/hip_bf16.h>
#include <stdint.h>

typedef __attribute__((ext_vector_type(16))) _Float16 v16h;
typedef __attribute__((ext_vector_type(8)))  float    v8f;

// ---------------- utility ----------------
__global__ void zero_f32(float* __restrict__ p, long long n) {
    long long i = (long long)blockIdx.x * blockDim.x + threadIdx.x;
    if (i < n) p[i] = 0.0f;
}

// deg[dst[e]] += 1  (atomics to a 400KB array: resolves in L2)
__global__ void count_deg(const int* __restrict__ dst, float* __restrict__ deg, int E) {
    int e = blockIdx.x * blockDim.x + threadIdx.x;
    if (e < E) atomicAdd(&deg[dst[e]], 1.0f);
}

// dinv = rsqrt(deg + 1) in place
__global__ void make_dinv(float* __restrict__ deg, int N) {
    int i = blockIdx.x * blockDim.x + threadIdx.x;
    if (i < N) deg[i] = rsqrtf(deg[i] + 1.0f);
}

// ---------------- weight pre-swizzle into WMMA B-fragment layout ----------------
// B fragment layout (16-bit dense B, KxN per 16x16x32 WMMA): lane l holds column
// n = l&15, elems 0..7 -> K = (l>>4)*8+e ; 8..15 -> K = 16+(l>>4)*8+(e-8).
// W1 [128,32] -> Bp1[kb(4)][tile(2)][lane(32)][e(16)] halfs  (8 KB)
__global__ void pack_W1(const float* __restrict__ W, _Float16* __restrict__ Bp) {
    int i = blockIdx.x * blockDim.x + threadIdx.x;      // 4*2*32*16 = 4096
    if (i >= 4096) return;
    int e  = i & 15;
    int l  = (i >> 4) & 31;
    int t  = (i >> 9) & 1;
    int kb = i >> 10;
    int k  = kb * 32 + (l >> 4) * 8 + (e & 7) + (e >> 3) * 16;
    int n  = t * 16 + (l & 15);
    Bp[i] = (_Float16)W[k * 32 + n];
}

// W2 [32,16] -> Bp2[lane(32)][e(16)] halfs  (1 KB)
__global__ void pack_W2(const float* __restrict__ W, _Float16* __restrict__ Bp) {
    int i = blockIdx.x * blockDim.x + threadIdx.x;      // 32*16 = 512
    if (i >= 512) return;
    int e = i & 15;
    int l = i >> 4;
    int k = (l >> 4) * 8 + (e & 7) + (e >> 3) * 16;
    int n = l & 15;
    Bp[i] = (_Float16)W[k * 16 + n];
}

// ---------------- GEMM1: h1[N,32] = x[N,128] @ W1[128,32] via WMMA f16 ----------------
// one wave per 16-row tile; 4 K-steps of 32; two 16-wide N tiles (cols 0-15, 16-31)
__global__ void gemm1_wmma(const float* __restrict__ x, const v16h* __restrict__ Bp,
                           float* __restrict__ h, int N, int NT) {
    int wave = threadIdx.x >> 5;
    int lane = threadIdx.x & 31;
    int tile = blockIdx.x * (blockDim.x >> 5) + wave;
    if (tile >= NT) return;                 // wave-uniform: EXEC stays all-1 for WMMA
    int m  = lane & 15;                     // A row within tile
    int kh = lane >> 4;                     // K-half selector
    int row = tile * 16 + m;
    if (row >= N) row = N - 1;              // clamp loads, keep EXEC uniform
    const float* xr = x + (long long)row * 128;

    v8f acc0 = {}; v8f acc1 = {};
    #pragma unroll
    for (int kb = 0; kb < 4; ++kb) {
        // A: 16-bit A 16x32 layout: lane m, elems 0..7 -> K = kh*8+e ; 8..15 -> K = 16+kh*8+e
        int kA = kb * 32 + kh * 8;
        v16h a;
        #pragma unroll
        for (int e = 0; e < 8; ++e) {
            a[e]     = (_Float16)xr[kA + e];
            a[8 + e] = (_Float16)xr[kA + 16 + e];
        }
        // B: pre-swizzled fragments, one 32-byte vector load per fragment
        v16h b0 = Bp[(kb * 2 + 0) * 32 + lane];
        v16h b1 = Bp[(kb * 2 + 1) * 32 + lane];
        acc0 = __builtin_amdgcn_wmma_f32_16x16x32_f16(false, a, false, b0,
                                                      (short)0, acc0, false, false);
        acc1 = __builtin_amdgcn_wmma_f32_16x16x32_f16(false, a, false, b1,
                                                      (short)0, acc1, false, false);
    }
    // C/D layout: lane col n=m, VGPR r -> row M = kh*8 + r
    #pragma unroll
    for (int r = 0; r < 8; ++r) {
        int gr = tile * 16 + kh * 8 + r;
        if (gr < N) {
            long long o = (long long)gr * 32;
            h[o + m]      = acc0[r];
            h[o + 16 + m] = acc1[r];
        }
    }
}

// ---------------- GEMM2: h2[N,16] = h1r[N,32] @ W2[32,16] (single WMMA / tile) ----------------
__global__ void gemm2_wmma(const float* __restrict__ hin, const v16h* __restrict__ Bp,
                           float* __restrict__ h, int N, int NT) {
    int wave = threadIdx.x >> 5;
    int lane = threadIdx.x & 31;
    int tile = blockIdx.x * (blockDim.x >> 5) + wave;
    if (tile >= NT) return;
    int m  = lane & 15;
    int kh = lane >> 4;
    int row = tile * 16 + m;
    if (row >= N) row = N - 1;
    const float* xr = hin + (long long)row * 32;

    v16h a;
    int kA = kh * 8;
    #pragma unroll
    for (int e = 0; e < 8; ++e) {
        a[e]     = (_Float16)xr[kA + e];
        a[8 + e] = (_Float16)xr[kA + 16 + e];
    }
    v16h b = Bp[lane];

    v8f acc = {};
    acc = __builtin_amdgcn_wmma_f32_16x16x32_f16(false, a, false, b,
                                                 (short)0, acc, false, false);
    #pragma unroll
    for (int r = 0; r < 8; ++r) {
        int gr = tile * 16 + kh * 8 + r;
        if (gr < N) h[(long long)gr * 16 + m] = acc[r];
    }
}

// ---------------- edge scatter: wave-per-edge, lane = feature ----------------
// agg[dst, f] += h[src, f] * dinv[src] * dinv[dst]   (F = 32)
__global__ void scatter32(const int* __restrict__ src, const int* __restrict__ dst,
                          const float* __restrict__ dinv, const float* __restrict__ h,
                          float* __restrict__ agg, int E) {
    long long t = (long long)blockIdx.x * blockDim.x + threadIdx.x;
    int e = (int)(t >> 5);
    int f = (int)(t & 31);
    if (e >= E) return;
    int s = src[e], d = dst[e];
    float w = dinv[s] * dinv[d];
    atomicAdd(&agg[(long long)d * 32 + f], h[(long long)s * 32 + f] * w);
}

// half-wave per edge (F = 16)
__global__ void scatter16(const int* __restrict__ src, const int* __restrict__ dst,
                          const float* __restrict__ dinv, const float* __restrict__ h,
                          float* __restrict__ agg, int E) {
    long long t = (long long)blockIdx.x * blockDim.x + threadIdx.x;
    int e = (int)(t >> 4);
    int f = (int)(t & 15);
    if (e >= E) return;
    int s = src[e], d = dst[e];
    float w = dinv[s] * dinv[d];
    atomicAdd(&agg[(long long)d * 16 + f], h[(long long)s * 16 + f] * w);
}

// ---------------- epilogues ----------------
// h1r[i] = relu(agg[i] + h1[i]*dinv(n)^2 + b1[f])   (in place over agg)
__global__ void post1(const float* __restrict__ h1, float* __restrict__ agg,
                      const float* __restrict__ dinv, const float* __restrict__ bias, int N) {
    long long i = (long long)blockIdx.x * blockDim.x + threadIdx.x;
    if (i >= (long long)N * 32) return;
    int n = (int)(i >> 5);
    int f = (int)(i & 31);
    float di = dinv[n];
    float v = agg[i] + h1[i] * di * di + bias[f];
    agg[i] = v > 0.0f ? v : 0.0f;
}

// out[i] = out[i] + h2[i]*dinv(n)^2 + b2[f]
__global__ void post2(float* __restrict__ out, const float* __restrict__ h2,
                      const float* __restrict__ dinv, const float* __restrict__ bias, int N) {
    long long i = (long long)blockIdx.x * blockDim.x + threadIdx.x;
    if (i >= (long long)N * 16) return;
    int n = (int)(i >> 4);
    int f = (int)(i & 15);
    float di = dinv[n];
    out[i] = out[i] + h2[i] * di * di + bias[f];
}

// ---------------- launch ----------------
extern "C" void kernel_launch(void* const* d_in, const int* in_sizes, int n_in,
                              void* d_out, int out_size, void* d_ws, size_t ws_size,
                              hipStream_t stream) {
    const float* x  = (const float*)d_in[0];   // [N,128]
    const int*   ei = (const int*)d_in[1];     // [2,E] (harness delivers integer as int32)
    const float* W1 = (const float*)d_in[2];   // [128,32]
    const float* b1 = (const float*)d_in[3];   // [32]
    const float* W2 = (const float*)d_in[4];   // [32,16]
    const float* b2 = (const float*)d_in[5];   // [16]
    float* out = (float*)d_out;                // [N,16]

    const int N = in_sizes[0] / 128;
    const int E = in_sizes[1] / 2;
    const int* src = ei;
    const int* dst = ei + E;

    // workspace layout (base is 256-byte aligned):
    //   Bp1: 4096 halfs (8 KB, 32B-aligned)  | Bp2: 512 halfs (1 KB, 32B-aligned)
    //   dinv: N f32 | h1: N*32 f32 | agg1(->h1r): N*32 f32 | h2: N*16 f32
    _Float16* Bp1 = (_Float16*)d_ws;
    _Float16* Bp2 = Bp1 + 4096;
    float* dinv = (float*)(Bp2 + 512);
    float* h1   = dinv + N;
    float* agg1 = h1 + (long long)N * 32;
    float* h2   = agg1 + (long long)N * 32;

    const int B = 256;
    const int NT = (N + 15) / 16;
    const int wavesPerBlk = B / 32;

    // zero accumulators (out is poisoned by harness: must re-zero every call)
    zero_f32<<<(N + B - 1) / B, B, 0, stream>>>(dinv, N);
    zero_f32<<<((long long)N * 32 + B - 1) / B, B, 0, stream>>>(agg1, (long long)N * 32);
    zero_f32<<<((long long)N * 16 + B - 1) / B, B, 0, stream>>>(out, (long long)N * 16);

    // pre-swizzle weights into WMMA B-fragment layout
    pack_W1<<<(4096 + B - 1) / B, B, 0, stream>>>(W1, Bp1);
    pack_W2<<<(512 + B - 1) / B, B, 0, stream>>>(W2, Bp2);

    // degrees -> dinv
    count_deg<<<(E + B - 1) / B, B, 0, stream>>>(dst, dinv, E);
    make_dinv<<<(N + B - 1) / B, B, 0, stream>>>(dinv, N);

    // layer 1
    gemm1_wmma<<<(NT + wavesPerBlk - 1) / wavesPerBlk, B, 0, stream>>>(
        x, (const v16h*)Bp1, h1, N, NT);
    {
        long long tot = (long long)E * 32;
        scatter32<<<(unsigned)((tot + B - 1) / B), B, 0, stream>>>(src, dst, dinv, h1, agg1, E);
    }
    post1<<<((long long)N * 32 + B - 1) / B, B, 0, stream>>>(h1, agg1, dinv, b1, N);

    // layer 2 (agg1 now holds relu'd hidden features)
    gemm2_wmma<<<(NT + wavesPerBlk - 1) / wavesPerBlk, B, 0, stream>>>(
        agg1, (const v16h*)Bp2, h2, N, NT);
    {
        long long tot = (long long)E * 16;
        scatter16<<<(unsigned)((tot + B - 1) / B), B, 0, stream>>>(src, dst, dinv, h2, out, E);
    }
    post2<<<((long long)N * 16 + B - 1) / B, B, 0, stream>>>(out, h2, dinv, b2, N);
}
